// Attention_31164282699813
// MI455X (gfx1250) — matrix operand order repaired and checked by reading, not compiled, not run on hardware
//
#include <hip/hip_runtime.h>

// ---------------------------------------------------------------------------
// CDNA5 / gfx1250 attention pipeline
//   - all GEMM-shaped math via v_wmma_f32_16x16x32_bf16 (f32 accumulate)
//   - shared tiles double-buffered in LDS, moved by the Tensor Data Mover
//     (TENSOR_LOAD_TO_LDS + s_wait_tensorcnt, CDNA5 ISA):
//     one descriptor per 64x64 bf16 tile, issued by wave 0 only; barrier
//     releases the block after the per-wave TENSORcnt drain.
// ---------------------------------------------------------------------------

typedef __bf16 v16bf __attribute__((ext_vector_type(16)));
typedef __bf16 v8bf  __attribute__((ext_vector_type(8)));
typedef float  v8f   __attribute__((ext_vector_type(8)));
typedef unsigned int v4u __attribute__((ext_vector_type(4)));
typedef int v4i __attribute__((ext_vector_type(4)));
typedef int v8i __attribute__((ext_vector_type(8)));

constexpr int BS  = 2;
constexpr int S   = 2048;
constexpr int D   = 2048;
constexpr int H   = 32;
constexpr int KVH = 8;
constexpr int HD  = 64;
constexpr int NREP = H / KVH;   // 4

// ---------------------------------------------------------------------------
// TDM: load a 2D tile (tile_d1 rows x tile_d0 contiguous elems, 2B elements)
// from global [tensor_d1 x tensor_d0, row pitch stride0] into LDS at lds_addr
// (rows stored contiguously). D# layout per CDNA5 ISA.
// ---------------------------------------------------------------------------
__device__ __forceinline__ void tdm_load_2d(unsigned lds_addr, const void* gaddr,
                                            unsigned tensor_d0, unsigned tensor_d1,
                                            unsigned stride0,
                                            unsigned tile_d0, unsigned tile_d1) {
  const unsigned long long ga = (unsigned long long)(uintptr_t)gaddr;
  v4u g0;
  g0[0] = 1u;                                            // count=1 valid descriptor
  g0[1] = lds_addr;                                      // lds_addr [63:32]
  g0[2] = (unsigned)(ga & 0xFFFFFFFFu);                  // global_addr [95:64]
  g0[3] = ((unsigned)(ga >> 32) & 0x01FFFFFFu)           // global_addr [120:96]
          | (2u << 30);                                  // type=2 ("image")
  v8i g1;
  g1[0] = (int)(1u << 16);                               // data_size=1 -> 2 bytes
  g1[1] = (int)((tensor_d0 & 0xFFFFu) << 16);            // tensor_dim0 [79:48]
  g1[2] = (int)((tensor_d0 >> 16) |
                ((tensor_d1 & 0xFFFFu) << 16));          // tensor_dim1 [111:80]
  g1[3] = (int)((tensor_d1 >> 16) |
                ((tile_d0 & 0xFFFFu) << 16));            // tile_dim0 [127:112]
  g1[4] = (int)(tile_d1 & 0xFFFFu);                      // tile_dim1 [143:128]
  g1[5] = (int)stride0;                                  // tensor_dim0_stride [207:160]
  g1[6] = 0;
  g1[7] = 0;
  v4i z4 = {0, 0, 0, 0};
#if defined(__clang_major__) && (__clang_major__ >= 23)
  v8i z8 = {0, 0, 0, 0, 0, 0, 0, 0};
  __builtin_amdgcn_tensor_load_to_lds(g0, g1, z4, z4, z8, 0);
#else
  __builtin_amdgcn_tensor_load_to_lds(g0, g1, z4, z4, 0);
#endif
}

#define WAIT_TENSOR(n) __builtin_amdgcn_s_wait_tensorcnt((short)(n))

// ---------------------------------------------------------------------------
// WMMA fragment layouts (CDNA5 ISA, wave32)
// A 16x32 (16-bit): lane l: row M=l%16; elems 0..7  -> K = e + (l>=16?8:0)
//                                 elems 8..15 -> K = 16 + (e-8) + (l>=16?8:0)
// B 32x16 (16-bit): lane l: col N=l%16; this code loads A and B both as K = e + (l>=16?16:0)
// C/D 16x16 (f32):  lane l: col N=l%16; vgpr j -> row M = j + (l>=16?8:0)
// ---------------------------------------------------------------------------

__device__ __forceinline__ v16bf pack16(v8bf lo, v8bf hi) {
  v16bf r;
#pragma unroll
  for (int i = 0; i < 8; ++i) { r[i] = lo[i]; r[i + 8] = hi[i]; }
  return r;
}

__device__ __forceinline__ v16bf load_frag_a(const __bf16* base, int ld) {
  const int lane = threadIdx.x & 31;
  const int row  = lane & 15;
  const __bf16* p = base + (size_t)row * ld + ((lane >= 16) ? 16 : 0);
  return pack16(*(const v8bf*)(p), *(const v8bf*)(p + 8));
}

__device__ __forceinline__ v16bf load_frag_b(const __bf16* base, int ld) {
  const int lane = threadIdx.x & 31;
  const int col  = lane & 15;
  const __bf16* p = base + (size_t)col * ld + ((lane >= 16) ? 16 : 0);
  return pack16(*(const v8bf*)(p), *(const v8bf*)(p + 8));
}

// ---------------------------------------------------------------------------
// f32 -> bf16 elementwise convert
// ---------------------------------------------------------------------------
__global__ void cvt_f32_bf16(const float* __restrict__ in,
                             __bf16* __restrict__ out, int n) {
  int i = blockIdx.x * blockDim.x + threadIdx.x;
  if (i < n) out[i] = (__bf16)in[i];
}

// ---------------------------------------------------------------------------
// C[M,N] (f32) = A[M,K] (bf16, row-major) * B[N,K]^T (bf16, row-major)
// Block = 128 threads = 4 waves; block tile 64x64, K-step 64.
// B tile (64x64 = 8KB) double-buffered in LDS, one TDM descriptor per tile.
// ---------------------------------------------------------------------------
__global__ __launch_bounds__(128) void gemm_bf16_nt(
    const __bf16* __restrict__ A, const __bf16* __restrict__ B,
    float* __restrict__ C, int M, int N, int K) {
  __shared__ __align__(16) __bf16 sB[2][64 * 64];   // 2 x 8KB

  const int wave = threadIdx.x >> 5;
  const int lane = threadIdx.x & 31;
  const int m0 = blockIdx.y * 64 + wave * 16;
  const int n0 = blockIdx.x * 64;
  const int col  = lane & 15;
  const int koff = (lane >= 16) ? 16 : 0;

  v8f acc[4];
#pragma unroll
  for (int t = 0; t < 4; ++t) {
    v8f z = {0.f,0.f,0.f,0.f,0.f,0.f,0.f,0.f};
    acc[t] = z;
  }

  const unsigned ldsbase = (unsigned)(uintptr_t)&sB[0][0];
  const __bf16* Btile = B + (size_t)n0 * K;         // 64 rows, pitch K
  const __bf16* Ab = A + (size_t)m0 * K;
  const int nIter = K >> 6;

  // Prologue: TDM tile 0 into buffer 0 (wave 0 only).
  if (wave == 0) tdm_load_2d(ldsbase, Btile, (unsigned)K, 64u, (unsigned)K, 64u, 64u);

  for (int it = 0; it < nIter; ++it) {
    const int cur = it & 1;
    if (wave == 0) {
      if (it + 1 < nIter) {
        tdm_load_2d(ldsbase + (unsigned)((cur ^ 1) * 8192),
                    Btile + ((it + 1) << 6), (unsigned)K, 64u, (unsigned)K, 64u, 64u);
        WAIT_TENSOR(1);   // in-order per-wave TDM: tile `it` has landed
      } else {
        WAIT_TENSOR(0);
      }
    }
    __syncthreads();

    if (it + 1 < nIter) __builtin_prefetch(Ab + ((it + 1) << 6), 0, 3);

#pragma unroll
    for (int sub = 0; sub < 2; ++sub) {
      v16bf a = load_frag_a(Ab + (it << 6) + sub * 32, K);
      v16bf bf[4];
#pragma unroll
      for (int t = 0; t < 4; ++t) {
        const int rl = t * 16 + col;
        v8bf lo = *(const v8bf*)&sB[cur][rl * 64 + sub * 32 + koff];
        v8bf hh = *(const v8bf*)&sB[cur][rl * 64 + sub * 32 + koff + 8];
        bf[t] = pack16(lo, hh);
      }
#pragma unroll
      for (int t = 0; t < 4; ++t)
        acc[t] = __builtin_amdgcn_wmma_f32_16x16x32_bf16(false, a, false, bf[t], (short)0, acc[t], false, false);
    }
    __syncthreads();
  }

  const int roff = (lane >= 16) ? 8 : 0;
  float* Cb = C + (size_t)(m0 + roff) * N + n0 + col;
#pragma unroll
  for (int t = 0; t < 4; ++t)
#pragma unroll
    for (int j = 0; j < 8; ++j)
      Cb[(size_t)j * N + 16 * t] = acc[t][j];
}

// ---------------------------------------------------------------------------
// RoPE + layout change + bf16 convert.
// Input  tf:  [b, s, nh, HD] f32   Output out: [b, nh, s, HD] bf16
// ---------------------------------------------------------------------------
__global__ void rope_cvt(const float* __restrict__ tf,
                         const float* __restrict__ fc,
                         __bf16* __restrict__ out, int nh, int total) {
  int i = blockIdx.x * blockDim.x + threadIdx.x;
  if (i >= total) return;
  int pr = i & 31;
  int t  = i >> 5;
  int h  = t % nh; t /= nh;
  int s  = t % S;
  int b  = t / S;
  const float* src = tf + ((((size_t)b * S + s) * nh + h) * HD) + pr * 2;
  float t0 = src[0], t1 = src[1];
  float cs = fc[(s * 32 + pr) * 2 + 0];
  float sn = fc[(s * 32 + pr) * 2 + 1];
  __bf16* dst = out + ((((size_t)b * nh + h) * S + s) * HD) + pr * 2;
  dst[0] = (__bf16)(t0 * cs - t1 * sn);
  dst[1] = (__bf16)(t0 * sn + t1 * cs);
}

// ---------------------------------------------------------------------------
// V: [b, s, g, HD] f32  ->  V^T: [b, g, HD, s] bf16
// ---------------------------------------------------------------------------
__global__ void v_cvt_transpose(const float* __restrict__ vf,
                                __bf16* __restrict__ vo, int total) {
  int i = blockIdx.x * blockDim.x + threadIdx.x;
  if (i >= total) return;
  int s  = i % S;  int t = i / S;
  int hd = t % HD; t /= HD;
  int g  = t % KVH;
  int b  = t / KVH;
  vo[i] = (__bf16)vf[(((size_t)b * S + s) * KVH + g) * HD + hd];
}

// ---------------------------------------------------------------------------
// Flash attention (transposed):  S^T = K*Q^T,  O^T = V^T*P^T.
// Block = (64-query tile, head, batch) = 128 threads = 4 waves.
// K-tile and V^T-tile (8KB each) double-buffered in LDS, two TDM
// descriptors per key-tile iteration (wave 0 issues, TENSORcnt drains).
// ---------------------------------------------------------------------------
__global__ __launch_bounds__(128) void attn_fa(
    const __bf16* __restrict__ qbf,   // [b, h, s, HD]
    const __bf16* __restrict__ kbf,   // [b, g, s, HD]
    const __bf16* __restrict__ vT,    // [b, g, HD, s]
    const int*    __restrict__ seq,   // [b, s]
    __bf16* __restrict__ outbf) {     // [b, s, H*HD]
  __shared__ __align__(16) __bf16 sK[2][64 * 64];   // [buf][key_local][hd]
  __shared__ __align__(16) __bf16 sV[2][64 * 64];   // [buf][hd_local][key_local]

  const int bx   = blockIdx.x;
  const int h    = blockIdx.y;
  const int b    = blockIdx.z;
  const int g    = h / NREP;
  const int wave = threadIdx.x >> 5;
  const int lane = threadIdx.x & 31;
  const int col  = lane & 15;
  const bool hi  = lane >= 16;
  const int aoff = hi ? 8 : 0;        // C/D row offset; A-frag K offset is 2*aoff
  const int q0   = bx * 64 + wave * 16;
  const int q    = q0 + col;
  const int sq   = seq[b * S + q];

  // Q^T B-fragments, resident for the whole kernel.
  const __bf16* qrow = qbf + (((size_t)b * H + h) * S + q0) * HD;
  v16bf qb0 = load_frag_b(qrow +  0, HD);
  v16bf qb1 = load_frag_b(qrow + 32, HD);

  v8f o[4];
#pragma unroll
  for (int t = 0; t < 4; ++t) {
    v8f z = {0.f,0.f,0.f,0.f,0.f,0.f,0.f,0.f};
    o[t] = z;
  }
  float m = -1e30f;
  float l_sum = 0.f;

  const __bf16* kbase = kbf + (((size_t)b * KVH + g) * S) * HD;
  const __bf16* vbase = vT  + (((size_t)b * KVH + g) * HD) * S;
  const float scale = 0.125f;   // 1/sqrt(64)

  const unsigned ldsK0 = (unsigned)(uintptr_t)&sK[0][0];
  const unsigned ldsV0 = (unsigned)(uintptr_t)&sV[0][0];

  // Prologue: TDM key-tile 0 into buffer 0 (wave 0 only).
  if (wave == 0) {
    tdm_load_2d(ldsK0, kbase, (unsigned)HD, 64u, (unsigned)HD, 64u, 64u);
    tdm_load_2d(ldsV0, vbase, (unsigned)S, (unsigned)HD, (unsigned)S, 64u, 64u);
  }

  for (int kt = 0; kt <= bx; ++kt) {
    const int cur = kt & 1;
    const int kstart = kt * 64;

    if (wave == 0) {
      if (kt < bx) {
        const unsigned add = (unsigned)((cur ^ 1) * 8192);
        const int knext = kstart + 64;
        tdm_load_2d(ldsK0 + add, kbase + (size_t)knext * HD,
                    (unsigned)HD, 64u, (unsigned)HD, 64u, 64u);
        tdm_load_2d(ldsV0 + add, vbase + knext,
                    (unsigned)S, (unsigned)HD, (unsigned)S, 64u, 64u);
        WAIT_TENSOR(2);   // in-order per-wave TDM: tile `kt` has landed
      } else {
        WAIT_TENSOR(0);
      }
    }
    __syncthreads();

    // ---- scores: S^T = K * Q^T ------------------------------------------
    v16bf ka[4][2];
#pragma unroll
    for (int kg = 0; kg < 4; ++kg) {
      const int rl = kg * 16 + col;
#pragma unroll
      for (int c = 0; c < 2; ++c) {
        v8bf lo = *(const v8bf*)&sK[cur][rl * 64 + c * 32 + 2 * aoff];
        v8bf hh = *(const v8bf*)&sK[cur][rl * 64 + c * 32 + 2 * aoff + 8];
        ka[kg][c] = pack16(lo, hh);
      }
    }
    float p[4][8];
    float tmax = -1e30f;
#pragma unroll
    for (int kg = 0; kg < 4; ++kg) {
      v8f sf = {0.f,0.f,0.f,0.f,0.f,0.f,0.f,0.f};
      sf = __builtin_amdgcn_wmma_f32_16x16x32_bf16(false, ka[kg][0], false, qb0, (short)0, sf, false, false);
      sf = __builtin_amdgcn_wmma_f32_16x16x32_bf16(false, ka[kg][1], false, qb1, (short)0, sf, false, false);
#pragma unroll
      for (int j = 0; j < 8; ++j) {
        const int key = kstart + kg * 16 + j + aoff;
        float v = sf[j] * scale;
        const bool ok = (q >= key) && (seq[b * S + key] == sq);
        v = ok ? v : -1e30f;
        p[kg][j] = v;
        tmax = fmaxf(tmax, v);
      }
    }

    // ---- online softmax (per-query stats are per-lane) ------------------
    tmax = fmaxf(tmax, __shfl_xor(tmax, 16));
    const float newm = fmaxf(m, tmax);
    const float r = __expf(m - newm);
    m = newm;
    l_sum *= r;
#pragma unroll
    for (int t = 0; t < 4; ++t)
#pragma unroll
      for (int j = 0; j < 8; ++j) o[t][j] *= r;

    float part = 0.f;
#pragma unroll
    for (int kg = 0; kg < 4; ++kg)
#pragma unroll
      for (int j = 0; j < 8; ++j) {
        p[kg][j] = __expf(p[kg][j] - m);
        part += p[kg][j];
      }
    part += __shfl_xor(part, 16);
    l_sum += part;

    // Rebuild P^T as B-fragments from C-layout fragments (half-wave swaps).
    v16bf pb[2];
#pragma unroll
    for (int c = 0; c < 2; ++c) {
#pragma unroll
      for (int j = 0; j < 8; ++j) {
        const float a0 = p[2 * c][j];
        const float a1 = p[2 * c + 1][j];
        const float x0 = __shfl_xor(a0, 16);
        const float x1 = __shfl_xor(a1, 16);
        pb[c][j]     = (__bf16)(hi ? x1 : a0);
        pb[c][j + 8] = (__bf16)(hi ? a1 : x0);
      }
    }

    // ---- O^T += V^T * P^T ----------------------------------------------
    v16bf va[4][2];
#pragma unroll
    for (int hg = 0; hg < 4; ++hg) {
      const int rl = hg * 16 + col;
#pragma unroll
      for (int c = 0; c < 2; ++c) {
        v8bf lo = *(const v8bf*)&sV[cur][rl * 64 + c * 32 + 2 * aoff];
        v8bf hh = *(const v8bf*)&sV[cur][rl * 64 + c * 32 + 2 * aoff + 8];
        va[hg][c] = pack16(lo, hh);
      }
    }
#pragma unroll
    for (int hg = 0; hg < 4; ++hg) {
      o[hg] = __builtin_amdgcn_wmma_f32_16x16x32_bf16(false, va[hg][0], false, pb[0], (short)0, o[hg], false, false);
      o[hg] = __builtin_amdgcn_wmma_f32_16x16x32_bf16(false, va[hg][1], false, pb[1], (short)0, o[hg], false, false);
    }
    __syncthreads();
  }

  const float inv = (l_sum > 0.f) ? (1.0f / l_sum) : 0.f;
  __bf16* dst = outbf + ((size_t)b * S + q) * (H * HD) + h * HD;
#pragma unroll
  for (int hg = 0; hg < 4; ++hg)
#pragma unroll
    for (int j = 0; j < 8; ++j) {
      const int hd = hg * 16 + j + aoff;
      dst[hd] = (__bf16)(o[hg][j] * inv);
    }
}

// ---------------------------------------------------------------------------
// Host-side orchestration
// ---------------------------------------------------------------------------
extern "C" void kernel_launch(void* const* d_in, const int* in_sizes, int n_in,
                              void* d_out, int out_size, void* d_ws, size_t ws_size,
                              hipStream_t stream) {
  (void)in_sizes; (void)n_in; (void)out_size; (void)ws_size;

  const float* x   = (const float*)d_in[0];   // [BS,S,D]
  const float* fc  = (const float*)d_in[1];   // [S,32,2]
  const int*   seq = (const int*)  d_in[2];   // [BS,S]
  const float* wq  = (const float*)d_in[3];   // [H*HD, D]
  const float* wk  = (const float*)d_in[4];   // [KVH*HD, D]
  const float* wv  = (const float*)d_in[5];   // [KVH*HD, D]
  const float* wo  = (const float*)d_in[6];   // [D, H*HD]
  float* out = (float*)d_out;                 // [BS,S,D]

  constexpr size_t N_X  = (size_t)BS * S * D;
  constexpr size_t N_WQ = (size_t)H * HD * D;
  constexpr size_t N_WK = (size_t)KVH * HD * D;
  constexpr size_t N_Q  = (size_t)BS * S * H * HD;
  constexpr size_t N_K  = (size_t)BS * S * KVH * HD;

  char* w = (char*)d_ws;
  size_t off = 0;
  auto carve = [&](size_t bytes) { char* p = w + off; off += (bytes + 255) & ~(size_t)255; return p; };

  __bf16* xbf  = (__bf16*)carve(N_X  * 2);
  __bf16* wqbf = (__bf16*)carve(N_WQ * 2);
  __bf16* wkbf = (__bf16*)carve(N_WK * 2);
  __bf16* wvbf = (__bf16*)carve(N_WK * 2);
  __bf16* wobf = (__bf16*)carve(N_WQ * 2);
  float*  qf   = (float*) carve(N_Q  * 4);
  float*  kf   = (float*) carve(N_K  * 4);
  float*  vf   = (float*) carve(N_K  * 4);
  __bf16* qbf  = (__bf16*)carve(N_Q  * 2);
  __bf16* kbf  = (__bf16*)carve(N_K  * 2);
  __bf16* vTbf = (__bf16*)carve(N_K  * 2);
  __bf16* aobf = (__bf16*)carve(N_Q  * 2);

  const int TB = 256;
  cvt_f32_bf16<<<(int)((N_X  + TB - 1) / TB), TB, 0, stream>>>(x,  xbf,  (int)N_X);
  cvt_f32_bf16<<<(int)((N_WQ + TB - 1) / TB), TB, 0, stream>>>(wq, wqbf, (int)N_WQ);
  cvt_f32_bf16<<<(int)((N_WK + TB - 1) / TB), TB, 0, stream>>>(wk, wkbf, (int)N_WK);
  cvt_f32_bf16<<<(int)((N_WK + TB - 1) / TB), TB, 0, stream>>>(wv, wvbf, (int)N_WK);
  cvt_f32_bf16<<<(int)((N_WQ + TB - 1) / TB), TB, 0, stream>>>(wo, wobf, (int)N_WQ);

  const int M = BS * S;           // 4096
  gemm_bf16_nt<<<dim3((H * HD) / 64, M / 64), 128, 0, stream>>>(xbf, wqbf, qf, M, H * HD, D);
  gemm_bf16_nt<<<dim3((KVH * HD) / 64, M / 64), 128, 0, stream>>>(xbf, wkbf, kf, M, KVH * HD, D);
  gemm_bf16_nt<<<dim3((KVH * HD) / 64, M / 64), 128, 0, stream>>>(xbf, wvbf, vf, M, KVH * HD, D);

  const int nq_rope = BS * S * H * 32;
  const int nk_rope = BS * S * KVH * 32;
  rope_cvt<<<(nq_rope + TB - 1) / TB, TB, 0, stream>>>(qf, fc, qbf, H,   nq_rope);
  rope_cvt<<<(nk_rope + TB - 1) / TB, TB, 0, stream>>>(kf, fc, kbf, KVH, nk_rope);
  v_cvt_transpose<<<(int)((N_K + TB - 1) / TB), TB, 0, stream>>>(vf, vTbf, (int)N_K);

  attn_fa<<<dim3(S / 64, H, BS), 128, 0, stream>>>(qbf, kbf, vTbf, seq, aobf);

  gemm_bf16_nt<<<dim3(D / 64, M / 64), 128, 0, stream>>>(aobf, wobf, out, M, D, H * HD);
}
